// CrossAttentionModule_8847632629857
// MI455X (gfx1250) — matrix operand order repaired and checked by reading, not compiled, not run on hardware
//
#include <hip/hip_runtime.h>
#include <stdint.h>

typedef __bf16 v16bf __attribute__((ext_vector_type(16)));
typedef float  v8f   __attribute__((ext_vector_type(8)));

union Frag {
  v16bf v;
  uint4 q[2];
};

__device__ __forceinline__ unsigned short f2bf(float f) {
  unsigned int u = __float_as_uint(f);
  u += 0x7FFFu + ((u >> 16) & 1u);   // round-to-nearest-even
  return (unsigned short)(u >> 16);
}

// ---------------------------------------------------------------------------
// Elementwise f32 -> bf16
// ---------------------------------------------------------------------------
__global__ void f32_to_bf16_kernel(const float* __restrict__ in,
                                   unsigned short* __restrict__ out, int n) {
  int i = blockIdx.x * blockDim.x + threadIdx.x;
  if (i < n) out[i] = f2bf(in[i]);
}

// in[K][N] f32 -> out[N][K] bf16   (weights pre-transposed so WMMA B-operand
// columns are contiguous 32B loads)
__global__ void transpose_w_bf16(const float* __restrict__ in,
                                 unsigned short* __restrict__ out,
                                 int K, int N) {
  int i = blockIdx.x * blockDim.x + threadIdx.x;
  if (i < K * N) {
    int n = i / K, k = i % K;
    out[i] = f2bf(in[(size_t)k * N + n]);
  }
}

// ---------------------------------------------------------------------------
// QKV projection GEMM (bf16 WMMA, f32 acc) with fused RoPE epilogue.
// MODE 0: Q  (rope + 1/sqrt(64) scale), store [b,h,s,64]
// MODE 1: K  (rope),                    store [b,h,s,64]
// MODE 2: V  (no rope),                 store transposed [b,h,64,s]
// Block = 128 threads (4 waves); wave -> 16 rows x 128 cols (8 WMMA tiles),
// K-loop step 32.  8 WMMAs per A-fragment load maximizes matrix-unit duty.
// ---------------------------------------------------------------------------
template <int MODE>
__global__ void qkv_gemm_rope(const unsigned short* __restrict__ X,   // [R][1024] bf16
                              const unsigned short* __restrict__ Wt,  // [1024][1024] bf16 (N-major)
                              unsigned short* __restrict__ Out,
                              int S) {                                // tokens per batch
  const int lane = threadIdx.x & 31;
  const int wave = threadIdx.x >> 5;
  const int hlf  = lane >> 4;   // half-wave (0/1)
  const int lm   = lane & 15;
  const int mbase = (blockIdx.x * 4 + wave) * 16;
  const int nbase = blockIdx.y * 128;

  v8f acc[8] = {};
  const unsigned short* Xr = X + (size_t)(mbase + lm) * 1024;

  for (int k0 = 0; k0 < 1024; k0 += 32) {
    Frag a;                                   // A 16x32: lane=row, K pairs split by half
    const int kA = k0 + hlf * 16;
    a.q[0] = *(const uint4*)(Xr + kA);
    a.q[1] = *(const uint4*)(Xr + kA + 8);
    if (k0 + 32 < 1024) __builtin_prefetch(Xr + kA + 32, 0, 3);  // global_prefetch_b8
#pragma unroll
    for (int t = 0; t < 8; ++t) {
      const int n = nbase + t * 16 + lm;      // B 32x16: lane=col, 16 contiguous K
      Frag b;
      const unsigned short* Wr = Wt + (size_t)n * 1024 + k0 + hlf * 16;
      b.q[0] = *(const uint4*)(Wr);
      b.q[1] = *(const uint4*)(Wr + 8);
      acc[t] = __builtin_amdgcn_wmma_f32_16x16x32_bf16(false, a.v, false, b.v,
                                                       (short)0, acc[t], false, false);
    }
  }

#pragma unroll
  for (int t = 0; t < 8; ++t) {
    const int n  = nbase + t * 16 + lm;
    const int h  = n >> 6;
    const int dh = n & 63;
    if (MODE == 2) {
#pragma unroll
      for (int r = 0; r < 8; ++r) {
        const int gr = mbase + r + 8 * hlf;   // C layout: lanes 0-15 rows 0-7, 16-31 rows 8-15
        const int b = gr / S, s = gr % S;
        Out[((size_t)((b * 16 + h) * 64 + dh)) * (size_t)S + s] = f2bf(acc[t][r]);
      }
    } else {
      const int   i   = dh >> 1;
      const float inv = __expf(-9.210340371976184f * (float)(2 * i) * (1.0f / 64.0f));
#pragma unroll
      for (int r = 0; r < 8; ++r) {
        const float x  = acc[t][r];
        const float xo = __shfl_xor(x, 1, 32); // RoPE partner column = adjacent lane
        const int gr = mbase + r + 8 * hlf;
        const int b = gr / S, s = gr % S;
        float sn, cs;
        __sincosf((float)s * inv, &sn, &cs);
        float res = (dh & 1) ? (x * cs + xo * sn) : (x * cs - xo * sn);
        if (MODE == 0) res *= 0.125f;          // 1/sqrt(Dh)
        Out[((size_t)(b * 16 + h) * (size_t)S + s) * 64 + dh] = f2bf(res);
      }
    }
  }
}

// ---------------------------------------------------------------------------
// Flash attention: one wave per (b, h, 16 query rows). kv chunks of 32.
// Q,K: [b,h,s,64] bf16; Vt: [b,h,64,skv] bf16; out: [b*1024+q][h*64+dh] bf16.
// ---------------------------------------------------------------------------
__global__ void __launch_bounds__(32)
attn_kernel(const unsigned short* __restrict__ Qm,
            const unsigned short* __restrict__ Km,
            const unsigned short* __restrict__ Vtm,
            const unsigned char*  __restrict__ mask,
            unsigned short* __restrict__ Om) {
  const int lane = threadIdx.x & 31;
  const int hlf  = lane >> 4;
  const int lm   = lane & 15;
  const int bh   = blockIdx.x >> 6;   // b*16+h
  const int qt   = blockIdx.x & 63;
  const int b    = bh >> 4;
  const int h    = bh & 15;
  const int qbase = qt * 16;

  __shared__ __align__(32) unsigned short Plds[16 * 32];

  Frag qa[2];                              // Q as A-operand, Dh=64 -> 2 frags
  const unsigned short* Qr = Qm + ((size_t)bh * 1024 + qbase + lm) * 64;
#pragma unroll
  for (int s = 0; s < 2; ++s) {
    const int kA = s * 32 + hlf * 16;
    qa[s].q[0] = *(const uint4*)(Qr + kA);
    qa[s].q[1] = *(const uint4*)(Qr + kA + 8);
  }

  float m[8], l[8];
  v8f O[4] = {};
#pragma unroll
  for (int r = 0; r < 8; ++r) { m[r] = -1e30f; l[r] = 0.0f; }

  const unsigned short* Kbase = Km + (size_t)bh * 2048 * 64;
  const unsigned short* Vbase = Vtm + (size_t)bh * 64 * 2048;

  for (int c = 0; c < 2048; c += 32) {
    if (c + 32 < 2048) {                    // prefetch next chunk (K rows + V cols)
      __builtin_prefetch(Kbase + (size_t)(c + 32 + lm) * 64 + hlf * 32, 0, 3);
      __builtin_prefetch(Vbase + (size_t)lm * 2048 + c + 32 + hlf * 16, 0, 3);
    }

    v8f S0 = {}, S1 = {};
#pragma unroll
    for (int s = 0; s < 2; ++s) {          // accumulate over Dh
      const int doff = s * 32 + hlf * 16;
      const v16bf kb0 = *(const v16bf*)(Kbase + (size_t)(c + lm) * 64 + doff);
      const v16bf kb1 = *(const v16bf*)(Kbase + (size_t)(c + 16 + lm) * 64 + doff);
      S0 = __builtin_amdgcn_wmma_f32_16x16x32_bf16(false, qa[s].v, false, kb0, (short)0, S0, false, false);
      S1 = __builtin_amdgcn_wmma_f32_16x16x32_bf16(false, qa[s].v, false, kb1, (short)0, S1, false, false);
    }

    if (!mask[b * 2048 + c + lm]) {
#pragma unroll
      for (int r = 0; r < 8; ++r) S0[r] = -1e9f;
    }
    if (!mask[b * 2048 + c + 16 + lm]) {
#pragma unroll
      for (int r = 0; r < 8; ++r) S1[r] = -1e9f;
    }

    float rmax[8], p0[8], p1[8], rs[8];
#pragma unroll
    for (int r = 0; r < 8; ++r) rmax[r] = fmaxf(S0[r], S1[r]);
#pragma unroll
    for (int off = 1; off < 16; off <<= 1)   // row lives in one 16-lane half
#pragma unroll
      for (int r = 0; r < 8; ++r) rmax[r] = fmaxf(rmax[r], __shfl_xor(rmax[r], off, 32));

#pragma unroll
    for (int r = 0; r < 8; ++r) {
      const float mn = fmaxf(m[r], rmax[r]);
      const float al = __expf(m[r] - mn);
      p0[r] = __expf(S0[r] - mn);
      p1[r] = __expf(S1[r] - mn);
      rs[r] = p0[r] + p1[r];
      m[r] = mn;
      l[r] *= al;
#pragma unroll
      for (int t = 0; t < 4; ++t) O[t][r] *= al;
    }
#pragma unroll
    for (int off = 1; off < 16; off <<= 1)
#pragma unroll
      for (int r = 0; r < 8; ++r) rs[r] += __shfl_xor(rs[r], off, 32);
#pragma unroll
    for (int r = 0; r < 8; ++r) l[r] += rs[r];

    // P (C-layout) -> LDS -> reload as A-operand layout
#pragma unroll
    for (int r = 0; r < 8; ++r) {
      Plds[(r + 8 * hlf) * 32 + lm]      = f2bf(p0[r]);
      Plds[(r + 8 * hlf) * 32 + 16 + lm] = f2bf(p1[r]);
    }
    __syncthreads();
    Frag pf;
    pf.q[0] = *(const uint4*)(&Plds[lm * 32 + hlf * 16]);
    pf.q[1] = *(const uint4*)(&Plds[lm * 32 + hlf * 16 + 8]);
    __syncthreads();

#pragma unroll
    for (int t = 0; t < 4; ++t) {
      const v16bf vb = *(const v16bf*)(Vbase + (size_t)(t * 16 + lm) * 2048 + c + hlf * 16);
      O[t] = __builtin_amdgcn_wmma_f32_16x16x32_bf16(false, pf.v, false, vb, (short)0, O[t], false, false);
    }
  }

#pragma unroll
  for (int t = 0; t < 4; ++t)
#pragma unroll
    for (int r = 0; r < 8; ++r) {
      const int q = qbase + r + 8 * hlf;
      Om[((size_t)(b * 1024 + q)) * 1024 + h * 64 + t * 16 + lm] = f2bf(O[t][r] / l[r]);
    }
}

// ---------------------------------------------------------------------------
// Output projection: attn(bf16) @ Wo + residual -> f32
// wave -> 16 rows x 128 cols (8 WMMA tiles)
// ---------------------------------------------------------------------------
__global__ void out_gemm_res(const unsigned short* __restrict__ A,    // [4096][1024] bf16
                             const unsigned short* __restrict__ Wt,   // [1024][1024] bf16 (N-major)
                             const float* __restrict__ resid,         // [4096][1024] f32
                             float* __restrict__ Out) {
  const int lane = threadIdx.x & 31;
  const int wave = threadIdx.x >> 5;
  const int hlf  = lane >> 4;
  const int lm   = lane & 15;
  const int mbase = (blockIdx.x * 4 + wave) * 16;
  const int nbase = blockIdx.y * 128;

  v8f acc[8] = {};
  const unsigned short* Ar = A + (size_t)(mbase + lm) * 1024;

  for (int k0 = 0; k0 < 1024; k0 += 32) {
    Frag a;
    const int kA = k0 + hlf * 16;
    a.q[0] = *(const uint4*)(Ar + kA);
    a.q[1] = *(const uint4*)(Ar + kA + 8);
    if (k0 + 32 < 1024) __builtin_prefetch(Ar + kA + 32, 0, 3);
#pragma unroll
    for (int t = 0; t < 8; ++t) {
      const int n = nbase + t * 16 + lm;
      Frag bfr;
      const unsigned short* Wr = Wt + (size_t)n * 1024 + k0 + hlf * 16;
      bfr.q[0] = *(const uint4*)(Wr);
      bfr.q[1] = *(const uint4*)(Wr + 8);
      acc[t] = __builtin_amdgcn_wmma_f32_16x16x32_bf16(false, a.v, false, bfr.v,
                                                       (short)0, acc[t], false, false);
    }
  }

#pragma unroll
  for (int t = 0; t < 8; ++t) {
    const int n = nbase + t * 16 + lm;
#pragma unroll
    for (int r = 0; r < 8; ++r) {
      const int gr = mbase + r + 8 * hlf;
      Out[(size_t)gr * 1024 + n] = acc[t][r] + resid[(size_t)gr * 1024 + n];
    }
  }
}

// ---------------------------------------------------------------------------
// Row LayerNorm: 1024-wide rows, eps = 1e-3
// ---------------------------------------------------------------------------
__global__ void layernorm_kernel(const float* __restrict__ x,
                                 const float* __restrict__ gamma,
                                 const float* __restrict__ beta,
                                 float* __restrict__ y) {
  const int row = blockIdx.x;
  const float* xr = x + (size_t)row * 1024;
  float s1 = 0.f, s2 = 0.f;
  for (int c = threadIdx.x; c < 1024; c += blockDim.x) {
    const float v = xr[c];
    s1 += v; s2 += v * v;
  }
  __shared__ float a1[256], a2[256];
  a1[threadIdx.x] = s1; a2[threadIdx.x] = s2;
  __syncthreads();
  for (int off = blockDim.x >> 1; off > 0; off >>= 1) {
    if (threadIdx.x < off) {
      a1[threadIdx.x] += a1[threadIdx.x + off];
      a2[threadIdx.x] += a2[threadIdx.x + off];
    }
    __syncthreads();
  }
  const float mu  = a1[0] * (1.0f / 1024.0f);
  const float var = a2[0] * (1.0f / 1024.0f) - mu * mu;
  const float rs  = rsqrtf(var + 1e-3f);
  for (int c = threadIdx.x; c < 1024; c += blockDim.x)
    y[(size_t)row * 1024 + c] = (xr[c] - mu) * rs * gamma[c] + beta[c];
}

// ---------------------------------------------------------------------------
// Launcher
// ---------------------------------------------------------------------------
extern "C" void kernel_launch(void* const* d_in, const int* in_sizes, int n_in,
                              void* d_out, int out_size, void* d_ws, size_t ws_size,
                              hipStream_t stream) {
  const float*         dq   = (const float*)d_in[0];          // [4,1024,1024]
  const float*         ekv  = (const float*)d_in[1];          // [4,2048,1024]
  const unsigned char* mask = (const unsigned char*)d_in[2];  // [4,2048] bool
  const float*         Wq   = (const float*)d_in[3];
  const float*         Wk   = (const float*)d_in[4];
  const float*         Wv   = (const float*)d_in[5];
  const float*         Wo   = (const float*)d_in[6];
  const float*         lgam = (const float*)d_in[7];
  const float*         lbet = (const float*)d_in[8];
  float*               out  = (float*)d_out;

  char* ws = (char*)d_ws;
  const size_t MB = 1u << 20;
  unsigned short* Xq   = (unsigned short*)(ws + 0 * MB);   // 8 MB
  unsigned short* Xkv  = (unsigned short*)(ws + 8 * MB);   // 16 MB
  unsigned short* Wqt  = (unsigned short*)(ws + 24 * MB);  // 2 MB
  unsigned short* Wkt  = (unsigned short*)(ws + 26 * MB);  // 2 MB
  unsigned short* Wvt  = (unsigned short*)(ws + 28 * MB);  // 2 MB
  unsigned short* Wot  = (unsigned short*)(ws + 30 * MB);  // 2 MB
  unsigned short* Qb   = (unsigned short*)(ws + 32 * MB);  // 8 MB   [b,h,1024,64]
  unsigned short* Kb   = (unsigned short*)(ws + 40 * MB);  // 16 MB  [b,h,2048,64]
  unsigned short* Vtb  = (unsigned short*)(ws + 56 * MB);  // 16 MB  [b,h,64,2048]
  unsigned short* Attn = (unsigned short*)(ws + 72 * MB);  // 8 MB   [4096][1024]
  float*          Xres = (float*)(ws + 80 * MB);           // 16 MB  [4096][1024]

  // 1) precision conversion + weight transposes
  f32_to_bf16_kernel<<<(4194304 + 255) / 256, 256, 0, stream>>>(dq,  Xq,  4194304);
  f32_to_bf16_kernel<<<(8388608 + 255) / 256, 256, 0, stream>>>(ekv, Xkv, 8388608);
  transpose_w_bf16<<<(1048576 + 255) / 256, 256, 0, stream>>>(Wq, Wqt, 1024, 1024);
  transpose_w_bf16<<<(1048576 + 255) / 256, 256, 0, stream>>>(Wk, Wkt, 1024, 1024);
  transpose_w_bf16<<<(1048576 + 255) / 256, 256, 0, stream>>>(Wv, Wvt, 1024, 1024);
  transpose_w_bf16<<<(1048576 + 255) / 256, 256, 0, stream>>>(Wo, Wot, 1024, 1024);

  // 2) QKV projections with fused RoPE / layout epilogues
  qkv_gemm_rope<0><<<dim3(64, 8),  128, 0, stream>>>(Xq,  Wqt, Qb,  1024);
  qkv_gemm_rope<1><<<dim3(128, 8), 128, 0, stream>>>(Xkv, Wkt, Kb,  2048);
  qkv_gemm_rope<2><<<dim3(128, 8), 128, 0, stream>>>(Xkv, Wvt, Vtb, 2048);

  // 3) flash attention: B*H*(Sq/16) = 4096 waves
  attn_kernel<<<4096, 32, 0, stream>>>(Qb, Kb, Vtb, mask, Attn);

  // 4) output projection + residual, then layernorm
  out_gemm_res<<<dim3(64, 8), 128, 0, stream>>>(Attn, Wot, dq, Xres);
  layernorm_kernel<<<4096, 256, 0, stream>>>(Xres, lgam, lbet, out);
}
